// BiLSTMPOSTagger_49967649521791
// MI455X (gfx1250) — matrix operand order repaired and checked by reading, not compiled, not run on hardware
//
#include <hip/hip_runtime.h>
#include <hip/hip_bf16.h>
#include <cstdint>
#include <cstddef>

// ---------------- types ----------------
typedef __attribute__((ext_vector_type(16))) __bf16 bf16x16;
typedef __attribute__((ext_vector_type(8)))  __bf16 bf16x8;
typedef __attribute__((ext_vector_type(4)))  __bf16 bf16x4;
typedef __attribute__((ext_vector_type(8)))  float  f32x8;
typedef __attribute__((ext_vector_type(4)))  unsigned int u32x4;
typedef __attribute__((ext_vector_type(8)))  int i32x8;
typedef __attribute__((ext_vector_type(4)))  int i32x4;

#define WMMA_BF16(a, b, c) \
  __builtin_amdgcn_wmma_f32_16x16x32_bf16(false, (a), false, (b), (short)0, (c), false, false)

// ---------------- problem dims ----------------
constexpr int Bz = 64;     // batch
constexpr int Ss = 512;    // seq len
constexpr int Ee = 1024;   // embed dim
constexpr int Hh = 1024;   // hidden
constexpr int G4 = 4096;   // 4*H
constexpr int Oo = 50;     // output classes
constexpr int NWG = 64;    // workgroups per direction in recurrent kernel

// TDM-padded LDS layout for weight tiles:
// DMA inserts 2 DWORDs (8B) of pad after every 256 DWORDs (1024B) of data.
// Row = 1024 bf16 = 2 chunks of 512 elems; chunk stride 516 elems (1032B),
// row stride 1032 elems (2064B) == 516 DW == 4 (mod 64) banks -> the 16 lanes
// of a B-fragment read hit disjoint bank groups.
constexpr int LCHUNK = 516;    // bf16 elems per padded 512-elem chunk
constexpr int LROW   = 1032;   // bf16 elems per padded 1024-elem row

// ---------------- workspace layout (bytes) ----------------
constexpr size_t SZ_X   = (size_t)Ss * Bz * Ee * 2;   // x bf16 [S][B][E]
constexpr size_t SZ_XG  = (size_t)Ss * Bz * G4 * 2;   // xgates bf16 [S][B][4H]
constexpr size_t SZ_H   = (size_t)Ss * Bz * Hh * 2;   // h bf16 [S][B][H]
constexpr size_t SZ_W   = (size_t)2 * G4 * Ee * 2;    // L=2 layers of [4H][E] bf16

constexpr size_t OFF_X    = 0;
constexpr size_t OFF_XGF  = OFF_X   + SZ_X;
constexpr size_t OFF_XGB  = OFF_XGF + SZ_XG;
constexpr size_t OFF_H0F  = OFF_XGB + SZ_XG;
constexpr size_t OFF_H0B  = OFF_H0F + SZ_H;
constexpr size_t OFF_H1F  = OFF_H0B + SZ_H;
constexpr size_t OFF_H1B  = OFF_H1F + SZ_H;
constexpr size_t OFF_WIHF = OFF_H1B + SZ_H;
constexpr size_t OFF_WHHF = OFF_WIHF + SZ_W;
constexpr size_t OFF_WIHB = OFF_WHHF + SZ_W;
constexpr size_t OFF_WHHB = OFF_WIHB + SZ_W;
constexpr size_t OFF_SYNC = OFF_WHHB + SZ_W;
constexpr size_t SZ_SYNC  = (size_t)2 * 2 * Ss * 4;   // [layer][dir][t] int

constexpr int SMEM_WTILE = 64 * LROW * 2;             // 132096 B padded 64x1024 bf16 tile
constexpr int SMEM_GEMM  = SMEM_WTILE;
constexpr int SMEM_REC   = SMEM_WTILE + 4 * 64 * 16 * 4 /*gates*/ + 64 * 16 * 4 /*c*/;

// ---------------- TDM: 2-D tile load Global->LDS with per-1024B padding ----------------
// rows x 2048B tile, contiguous rows in memory, data_size=8B, pad 8B per 1024B.
__device__ __forceinline__ void tdm_load_wtile(unsigned int lds_off, const void* gptr,
                                               unsigned int rows) {
  uint64_t ga = (uint64_t)(uintptr_t)gptr;
  u32x4 g0;
  g0[0] = 1u;                                   // count=1 (valid descriptor)
  g0[1] = lds_off;                              // lds_addr
  g0[2] = (unsigned int)ga;                     // global_addr[31:0]
  g0[3] = (unsigned int)(ga >> 32) | (2u << 30);// global_addr[56:32] | type=2
  i32x8 g1;
  g1[0] = (int)((3u << 16)     // data_size = 8B
              | (1u << 20)     // pad_enable
              | (7u << 22)     // pad_interval = 256 DWORDs (1024B)
              | (1u << 25));   // pad_amount   = 2 DWORDs (8B)
  g1[1] = (int)(256u << 16);                    // tensor_dim0 = 256 (8B units)
  g1[2] = (int)((rows & 0xffffu) << 16);        // tensor_dim0 hi | tensor_dim1 = rows
  g1[3] = (int)(256u << 16);                    // tensor_dim1 hi | tile_dim0 = 256
  g1[4] = (int)(rows & 0xffffu);                // tile_dim1 = rows | tile_dim2 = 0
  g1[5] = 256;                                  // tensor_dim0_stride = 256 (contiguous)
  g1[6] = 0;
  g1[7] = 0;
  i32x4 z4 = {0, 0, 0, 0};                      // 2-D tensor: groups 2/3 unused
  i32x8 z8 = {0, 0, 0, 0, 0, 0, 0, 0};
  __builtin_amdgcn_tensor_load_to_lds(g0, g1, z4, z4, z8, 0);
}

// padded-LDS element address for (row, k)
__device__ __forceinline__ const __bf16* wtile_ptr(const __bf16* base, int row, int k) {
  return base + (size_t)row * LROW + (k >> 9) * LCHUNK + (k & 511);
}

// ---------------- f32 -> bf16 convert ----------------
__global__ void cvt_f32_bf16_kernel(const float* __restrict__ in,
                                    __bf16* __restrict__ out, int n4) {
  int i = blockIdx.x * blockDim.x + threadIdx.x;
  if (i < n4) {
    float4 v = ((const float4*)in)[i];
    bf16x4 o;
    o[0] = (__bf16)v.x; o[1] = (__bf16)v.y; o[2] = (__bf16)v.z; o[3] = (__bf16)v.w;
    ((bf16x4*)out)[i] = o;
  }
}

// ---------------- embedding gather: x[t][b][:] = bf16(emb[src[b][t]]) ----------------
__global__ void embed_kernel(const int* __restrict__ src,
                             const float* __restrict__ emb,
                             __bf16* __restrict__ x) {
  int blk = blockIdx.x;            // t*B + b
  int t = blk >> 6;
  int b = blk & 63;
  int row = src[b * Ss + t];
  const float4* e = (const float4*)(emb + (size_t)row * Ee);
  bf16x4* o = (bf16x4*)(x + (size_t)blk * Ee);
  int i = threadIdx.x;             // 256 threads * 4 floats = 1024
  float4 v = e[i];
  bf16x4 w;
  w[0] = (__bf16)v.x; w[1] = (__bf16)v.y; w[2] = (__bf16)v.z; w[3] = (__bf16)v.w;
  o[i] = w;
}

// ---------------- input projection GEMM: xg = x @ Wih.T + bias (bf16 out) ----------------
// M = S*B (row = t*B+b), N = 4096, K = 1024.  reverse!=0: read x at (S-1-t, b).
// B-block (64 cols x 1024 K of Wih) staged to LDS once per WG via TDM.
__global__ __launch_bounds__(256)
void gemm_xw_kernel(const __bf16* __restrict__ x, const __bf16* __restrict__ w,
                    const float* __restrict__ bias, __bf16* __restrict__ xg,
                    int reverse) {
  const int m0 = blockIdx.x * 64;
  const int n0 = blockIdx.y * 64;
  extern __shared__ char smem[];
  const __bf16* wlds = (const __bf16*)smem;          // padded [64][1024]

  if (threadIdx.x < 32) {
    tdm_load_wtile(__builtin_amdgcn_groupstaticsize(), w + (size_t)n0 * Ee, 64);
    __builtin_amdgcn_s_wait_tensorcnt(0);
  }
  __syncthreads();

  const int wv = threadIdx.x >> 5, lane = threadIdx.x & 31;
  const int r = lane & 15, half = lane >> 4;
  const int nt = wv & 3, mtb = (wv >> 2) * 2;
  const int nrow = nt * 16 + r;                      // local B row (= output col - n0)
  const int ncol = n0 + nrow;

  const __bf16* ap[2];
  for (int i = 0; i < 2; i++) {
    int mrow = m0 + (mtb + i) * 16 + r;
    int t = mrow >> 6, b = mrow & 63;
    int sr = reverse ? ((Ss - 1 - t) * Bz + b) : mrow;
    ap[i] = x + (size_t)sr * Ee;
  }

  f32x8 acc0 = {}, acc1 = {};
  for (int k0 = 0; k0 < Ee; k0 += 32) {
    __builtin_prefetch(ap[0] + k0 + 512, 0, 0);
    union { bf16x16 v; bf16x8 h2[2]; } fa0, fa1;
    union { bf16x16 v; bf16x4 q[4]; } fb;
    const __bf16* bp = wtile_ptr(wlds, nrow, k0 + half * 16);
    fb.q[0] = *(const bf16x4*)(bp);
    fb.q[1] = *(const bf16x4*)(bp + 4);
    fb.q[2] = *(const bf16x4*)(bp + 8);
    fb.q[3] = *(const bf16x4*)(bp + 12);
    fa0.h2[0] = *(const bf16x8*)(ap[0] + k0 + half * 16);
    fa0.h2[1] = *(const bf16x8*)(ap[0] + k0 + half * 16 + 8);
    fa1.h2[0] = *(const bf16x8*)(ap[1] + k0 + half * 16);
    fa1.h2[1] = *(const bf16x8*)(ap[1] + k0 + half * 16 + 8);
    acc0 = WMMA_BF16(fa0.v, fb.v, acc0);
    acc1 = WMMA_BF16(fa1.v, fb.v, acc1);
  }
  float bb = bias[ncol];
  for (int i = 0; i < 8; i++) {
    int mr0 = m0 + mtb * 16 + half * 8 + i;
    int mr1 = m0 + (mtb + 1) * 16 + half * 8 + i;
    xg[(size_t)mr0 * G4 + ncol] = (__bf16)(acc0[i] + bb);
    xg[(size_t)mr1 * G4 + ncol] = (__bf16)(acc1[i] + bb);
  }
}

// ---------------- persistent recurrent LSTM layer (both directions) ----------------
// grid: (NWG, 2).  Each WG owns 16 hidden units (64 gate columns), keeps its
// Whh slice in LDS (TDM-staged, padded) and c-state in LDS across all 512
// timesteps.  Per step: gates = xg[t] + h_{t-1} @ WhhSlice.T (64x64x1024 WMMA),
// gate nonlinearity, h written to global, grid-step barrier via atomics.
__global__ __launch_bounds__(256)
void lstm_rec_kernel(const __bf16* __restrict__ xg_f, const __bf16* __restrict__ xg_b,
                     const __bf16* __restrict__ whh_f, const __bf16* __restrict__ whh_b,
                     __bf16* __restrict__ hout_f, __bf16* __restrict__ hout_b,
                     int* __restrict__ sync) {
  const int wg  = blockIdx.x;   // hidden-unit slice [wg*16, wg*16+16)
  const int dir = blockIdx.y;   // 0 = fwd, 1 = bwd (processed order)
  const __bf16* xg   = dir ? xg_b  : xg_f;
  const __bf16* whh  = dir ? whh_b : whh_f;
  __bf16*       hout = dir ? hout_b : hout_f;
  int* sy = sync + dir * Ss;

  extern __shared__ char smem[];
  const __bf16* wlds = (const __bf16*)smem;              // padded [64][1024]
  float* gat = (float*)(smem + SMEM_WTILE);              // [4][64][16]
  float* cst = gat + 4 * 64 * 16;                        // [64][16]

  const int tid = threadIdx.x;

  // Stage this WG's 64 Whh rows: 4 gate blocks of 16 contiguous rows each.
  if (tid < 32) {
    unsigned int lbase = __builtin_amdgcn_groupstaticsize();
    for (int g = 0; g < 4; g++)
      tdm_load_wtile(lbase + (unsigned int)(g * 16 * LROW * 2),
                     whh + ((size_t)(g * Hh + wg * 16)) * Hh, 16);
    __builtin_amdgcn_s_wait_tensorcnt(0);
  }
  for (int i = tid; i < 64 * 16; i += 256) cst[i] = 0.f;
  __syncthreads();

  const int wv = tid >> 5, lane = tid & 31;
  const int r = lane & 15, half = lane >> 4;
  const int nt = wv & 3;             // gate index for this wave
  const int mtb = (wv >> 2) * 2;     // batch-tile pair
  const int colg = nt * Hh + wg * 16 + r;               // column in xg
  const int nrow = nt * 16 + r;                         // local Whh row

  for (int t = 0; t < Ss; t++) {
    // init accumulators from precomputed input gates (bias folded in)
    f32x8 acc0, acc1;
    {
      const __bf16* xrow = xg + (size_t)t * Bz * G4;
      for (int i = 0; i < 8; i++) {
        int b0 = mtb * 16 + half * 8 + i;
        int b1 = (mtb + 1) * 16 + half * 8 + i;
        acc0[i] = (float)xrow[(size_t)b0 * G4 + colg];
        acc1[i] = (float)xrow[(size_t)b1 * G4 + colg];
      }
    }
    if (t > 0) {
      const __bf16* hprev = hout + (size_t)(t - 1) * Bz * Hh;
      const __bf16* a0p = hprev + (size_t)(mtb * 16 + r) * Hh;
      const __bf16* a1p = hprev + (size_t)((mtb + 1) * 16 + r) * Hh;
      for (int k0 = 0; k0 < Hh; k0 += 32) {
        union { bf16x16 v; bf16x8 h2[2]; } fa0, fa1;
        union { bf16x16 v; bf16x4 q[4]; } fb;
        const __bf16* bp = wtile_ptr(wlds, nrow, k0 + half * 16);
        fb.q[0] = *(const bf16x4*)(bp);
        fb.q[1] = *(const bf16x4*)(bp + 4);
        fb.q[2] = *(const bf16x4*)(bp + 8);
        fb.q[3] = *(const bf16x4*)(bp + 12);
        fa0.h2[0] = *(const bf16x8*)(a0p + k0 + half * 16);
        fa0.h2[1] = *(const bf16x8*)(a0p + k0 + half * 16 + 8);
        fa1.h2[0] = *(const bf16x8*)(a1p + k0 + half * 16);
        fa1.h2[1] = *(const bf16x8*)(a1p + k0 + half * 16 + 8);
        acc0 = WMMA_BF16(fa0.v, fb.v, acc0);
        acc1 = WMMA_BF16(fa1.v, fb.v, acc1);
      }
    }
    // stage gates to LDS: gat[gate][b][j]
    float* gslot = gat + nt * (64 * 16);
    for (int i = 0; i < 8; i++) {
      int b0 = mtb * 16 + half * 8 + i;
      int b1 = (mtb + 1) * 16 + half * 8 + i;
      gslot[b0 * 16 + r] = acc0[i];
      gslot[b1 * 16 + r] = acc1[i];
    }
    __syncthreads();
    // combine gates -> c, h
    for (int it = 0; it < 4; it++) {
      int idx = tid + it * 256;        // b*16 + j
      int b = idx >> 4, j = idx & 15;
      float gi = gat[0 * 1024 + idx];
      float gf = gat[1 * 1024 + idx];
      float gg = gat[2 * 1024 + idx];
      float go = gat[3 * 1024 + idx];
      float c  = cst[idx];
      float si = 1.f / (1.f + __expf(-gi));
      float sf = 1.f / (1.f + __expf(-gf));
      float so = 1.f / (1.f + __expf(-go));
      float cn = sf * c + si * tanhf(gg);
      float h  = so * tanhf(cn);
      cst[idx] = cn;
      hout[(size_t)t * Bz * Hh + (size_t)b * Hh + wg * 16 + j] = (__bf16)h;
    }
    __threadfence();
    __syncthreads();
    if (tid == 0) {
      __hip_atomic_fetch_add(&sy[t], 1, __ATOMIC_RELEASE, __HIP_MEMORY_SCOPE_AGENT);
      while (__hip_atomic_load(&sy[t], __ATOMIC_ACQUIRE, __HIP_MEMORY_SCOPE_AGENT) < NWG)
        __builtin_amdgcn_s_sleep(2);
    }
    __syncthreads();
  }
}

// ---------------- output projection: out[b][s][o] = [hf|hb_rev] @ Wout.T + bout ----------------
__global__ void out_proj_kernel(const __bf16* __restrict__ h1f,
                                const __bf16* __restrict__ h1b,
                                const float* __restrict__ Wout,
                                const float* __restrict__ bout,
                                float* __restrict__ out) {
  int blk = blockIdx.x;          // b*S + s
  int b = blk >> 9;
  int s = blk & 511;
  int o = threadIdx.x;
  if (o >= Oo) return;
  const __bf16* hf = h1f + ((size_t)s * Bz + b) * Hh;
  const __bf16* hb = h1b + ((size_t)(Ss - 1 - s) * Bz + b) * Hh;  // un-reverse
  const float* w = Wout + (size_t)o * (2 * Hh);
  float acc = bout[o];
  for (int k = 0; k < Hh; k++) acc += (float)hf[k] * w[k];
  for (int k = 0; k < Hh; k++) acc += (float)hb[k] * w[Hh + k];
  out[((size_t)b * Ss + s) * Oo + o] = acc;
}

// ---------------- host launcher ----------------
extern "C" void kernel_launch(void* const* d_in, const int* in_sizes, int n_in,
                              void* d_out, int out_size, void* d_ws, size_t ws_size,
                              hipStream_t stream) {
  const int*   src   = (const int*)  d_in[0];
  const float* emb   = (const float*)d_in[1];
  const float* WihF  = (const float*)d_in[2];
  const float* WhhF  = (const float*)d_in[3];
  const float* bF    = (const float*)d_in[4];
  const float* WihB  = (const float*)d_in[5];
  const float* WhhB  = (const float*)d_in[6];
  const float* bB    = (const float*)d_in[7];
  const float* Wout  = (const float*)d_in[8];
  const float* bout  = (const float*)d_in[9];
  float* out = (float*)d_out;
  char*  ws  = (char*)d_ws;

  __bf16* xbf   = (__bf16*)(ws + OFF_X);
  __bf16* xgF   = (__bf16*)(ws + OFF_XGF);
  __bf16* xgB   = (__bf16*)(ws + OFF_XGB);
  __bf16* h0f   = (__bf16*)(ws + OFF_H0F);
  __bf16* h0b   = (__bf16*)(ws + OFF_H0B);
  __bf16* h1f   = (__bf16*)(ws + OFF_H1F);
  __bf16* h1b   = (__bf16*)(ws + OFF_H1B);
  __bf16* wihf  = (__bf16*)(ws + OFF_WIHF);
  __bf16* whhf  = (__bf16*)(ws + OFF_WHHF);
  __bf16* wihb  = (__bf16*)(ws + OFF_WIHB);
  __bf16* whhb  = (__bf16*)(ws + OFF_WHHB);
  int*    sync  = (int*)   (ws + OFF_SYNC);

  (void)in_sizes; (void)n_in; (void)out_size; (void)ws_size;

  (void)hipFuncSetAttribute((const void*)gemm_xw_kernel,
                            hipFuncAttributeMaxDynamicSharedMemorySize, SMEM_GEMM);
  (void)hipFuncSetAttribute((const void*)lstm_rec_kernel,
                            hipFuncAttributeMaxDynamicSharedMemorySize, SMEM_REC);

  // reset step barriers (both layers)
  (void)hipMemsetAsync(sync, 0, SZ_SYNC, stream);

  // convert weights fp32 -> bf16 (each is L*4H*E = 8388608 elems)
  {
    const int n4 = (2 * G4 * Ee) / 4;        // 2097152
    const int blocks = n4 / 256;             // 8192
    cvt_f32_bf16_kernel<<<blocks, 256, 0, stream>>>(WihF, wihf, n4);
    cvt_f32_bf16_kernel<<<blocks, 256, 0, stream>>>(WhhF, whhf, n4);
    cvt_f32_bf16_kernel<<<blocks, 256, 0, stream>>>(WihB, wihb, n4);
    cvt_f32_bf16_kernel<<<blocks, 256, 0, stream>>>(WhhB, whhb, n4);
  }

  // embedding gather -> x bf16 [S][B][E]
  embed_kernel<<<Ss * Bz, 256, 0, stream>>>(src, emb, xbf);

  dim3 ggrid(Ss * Bz / 64, G4 / 64);   // (512, 64)

  // ---- layer 0 ----
  gemm_xw_kernel<<<ggrid, 256, SMEM_GEMM, stream>>>(xbf, wihf, bF, xgF, 0);
  gemm_xw_kernel<<<ggrid, 256, SMEM_GEMM, stream>>>(xbf, wihb, bB, xgB, 1);
  lstm_rec_kernel<<<dim3(NWG, 2), 256, SMEM_REC, stream>>>(
      xgF, xgB, whhf, whhb, h0f, h0b, sync);

  // ---- layer 1 (inputs are layer-0 hidden states, already in processed order) ----
  gemm_xw_kernel<<<ggrid, 256, SMEM_GEMM, stream>>>(h0f, wihf + (size_t)G4 * Ee, bF + G4, xgF, 0);
  gemm_xw_kernel<<<ggrid, 256, SMEM_GEMM, stream>>>(h0b, wihb + (size_t)G4 * Ee, bB + G4, xgB, 0);
  lstm_rec_kernel<<<dim3(NWG, 2), 256, SMEM_REC, stream>>>(
      xgF, xgB, whhf + (size_t)G4 * Ee, whhb + (size_t)G4 * Ee, h1f, h1b, sync + 2 * Ss);

  // ---- output projection ----
  out_proj_kernel<<<Bz * Ss, 64, 0, stream>>>(h1f, h1b, Wout, bout, out);
}
